// StyleBlock_12601434046946
// MI455X (gfx1250) — matrix operand order repaired and checked by reading, not compiled, not run on hardware
//
#include <hip/hip_runtime.h>

#define SQRT2F 1.41421356237309515f

typedef __bf16 bf16x16 __attribute__((ext_vector_type(16)));
typedef float  f32x8   __attribute__((ext_vector_type(8)));

constexpr int B_ = 16, CIN = 512, COUT = 512, H_ = 64, W_ = 64, LAT = 512;

__device__ __forceinline__ unsigned short f2bf(float f) {
  union { float f; unsigned int u; } v; v.f = f;
  unsigned int u = v.u;
  u += 0x7FFFu + ((u >> 16) & 1u);   // round-to-nearest-even
  return (unsigned short)(u >> 16);
}

__device__ __forceinline__ float lrelu_s2(float v) {
  return (v > 0.f ? v : 0.2f * v) * SQRT2F;
}

// ---- 1. style[b,c] = lrelu(latent @ (W/sqrt(512)).T + bias)*sqrt2 ----------
__global__ void k_style(const float* __restrict__ latent,
                        const float* __restrict__ aw,
                        const float* __restrict__ ab,
                        float* __restrict__ style) {
  int idx = blockIdx.x * 256 + threadIdx.x;            // B*CIN = 8192
  if (idx >= B_ * CIN) return;
  int c = idx & (CIN - 1);
  int b = idx >> 9;
  const float wmul = 0.0441941738241592f;              // 1/sqrt(512)
  const float* lr = latent + b * LAT;
  const float* wr = aw + c * LAT;
  float s = 0.f;
  for (int l = 0; l < LAT; ++l) s += lr[l] * wr[l];
  style[idx] = lrelu_s2(s * wmul + ab[c]);
}

// ---- 2a. wsq[o,i] = sum_tap conv_weight^2 ---------------------------------
__global__ void k_wsq(const float* __restrict__ cw, float* __restrict__ wsq) {
  int idx = blockIdx.x * 256 + threadIdx.x;            // 512*512
  if (idx >= COUT * CIN) return;
  const float* p = cw + idx * 9;
  float s = 0.f;
#pragma unroll
  for (int t = 0; t < 9; ++t) s += p[t] * p[t];
  wsq[idx] = s;
}

// ---- 2b. demod[b,o] = rsqrt(sum_i style^2 * wsq + 1e-8) -------------------
__global__ void k_demod(const float* __restrict__ style,
                        const float* __restrict__ wsq,
                        float* __restrict__ demod) {
  int idx = blockIdx.x * 256 + threadIdx.x;            // B*COUT = 8192
  if (idx >= B_ * COUT) return;
  int o = idx & (COUT - 1);
  int b = idx >> 9;
  const float* sr = style + b * CIN;
  const float* wr = wsq + o * CIN;
  float acc = 0.f;
  for (int i = 0; i < CIN; ++i) { float s = sr[i]; acc += s * s * wr[i]; }
  demod[idx] = rsqrtf(acc + 1e-8f);
}

// ---- 3. x -> bf16 with noise injection ------------------------------------
__global__ void k_xh(const float* __restrict__ x,
                     const float* __restrict__ noise,
                     const float* __restrict__ nwp,
                     unsigned short* __restrict__ xh) {
  int idx = blockIdx.x * 256 + threadIdx.x;            // 16*512*4096
  int hw = idx & 4095;
  int b  = idx >> 21;                                  // idx / (512*4096)
  float v = x[idx] + nwp[0] * noise[b * 4096 + hw];
  xh[idx] = f2bf(v);
}

// ---- 4. pack modulated bf16 weights in WMMA A-fragment order --------------
// frag = ((((b*32 + coutBlk)*3 + kh)*16 + cb)*3 + kw); 512 bf16 per frag,
// laid out lane-major: elem = frag*512 + lane*16 + j.
// A 16x32 bf16 layout: m = lane&15, kofs = (lane>=16)?8:0, k(j)=kofs+(j<8?j:j+8)
__global__ void k_pack(const float* __restrict__ cw,
                       const float* __restrict__ style,
                       const float* __restrict__ demod,
                       unsigned short* __restrict__ wfrag) {
  int idx = blockIdx.x * 256 + threadIdx.x;            // 16*32*144*512 = 37748736
  int j    = idx & 15;
  int lane = (idx >> 4) & 31;
  int frag = idx >> 9;
  int kw = frag % 3;  int t = frag / 3;
  int cb = t & 15;    t >>= 4;
  int kh = t % 3;     t /= 3;
  int cblk = t & 31;
  int b    = t >> 5;
  int m    = lane & 15;
  int kofs = (lane & 16) >> 1;                          // 0 or 8
  int k    = kofs + (j < 8 ? j : j + 8);
  int cout = cblk * 16 + m;
  int ci   = cb * 32 + k;
  float v = cw[(cout * CIN + ci) * 9 + kh * 3 + kw]
          * style[b * CIN + ci] * demod[b * COUT + cout];
  wfrag[idx] = f2bf(v);
}

// ---- 5. implicit-GEMM conv with WMMA --------------------------------------
// Block: 1 batch x 1 output row x 128 couts. 8 waves = 8 cout subtiles of 16.
// Wave: 16 couts x 64 pixels = 4 accumulators of 16x16 -> 4 WMMAs per A load.
// Per stage: prefetch all 3 A fragments (global), then per kw batch the 4 B
// fragment loads (LDS) ahead of the 4 WMMAs so dscnt waits can be staggered.
__global__ void __launch_bounds__(256) k_conv(
    const unsigned short* __restrict__ xh,
    const unsigned short* __restrict__ wfrag,
    const float* __restrict__ bias,
    float* __restrict__ out) {
  const int cb128 = blockIdx.x;                         // 0..3  cout block-of-128
  const int h     = blockIdx.y;                         // 0..63 output row
  const int b     = blockIdx.z;                         // 0..15 batch
  const int tid   = threadIdx.x;
  const int lane  = tid & 31;
  const int wave  = tid >> 5;                           // 0..7
  const int coutBlk = cb128 * 8 + wave;                 // 0..31 (16-wide)
  const int n  = lane & 15;                             // B-matrix column
  const int kb = (lane & 16) >> 1;                      // B k-offset: 0 or 8

  // col-major input stage: 66 cols x (32ch*2B + 16B pad); pad to 68 cols
  __shared__ __align__(16) unsigned short xs[68 * 40];

  // halo columns (x-col -1 and 64) are always zero-padding: write once
  if (tid < 64) {
    int ch  = tid >> 1;
    int col = (tid & 1) ? 65 : 0;
    xs[col * 40 + ch] = 0;
  }

  f32x8 acc[4] = {{}, {}, {}, {}};

  const unsigned short* xb = xh + b * CIN * (H_ * W_);
  const int sch = tid >> 3;                             // staging channel 0..31
  const int scg = tid & 7;                              // staging col-group 0..7

  for (int kh = 0; kh < 3; ++kh) {
    const int row = h + kh - 1;
    const bool rowOK = (unsigned)row < 64u;             // uniform scalar branch
    for (int cbk = 0; cbk < 16; ++cbk) {
      __syncthreads();
      // stage 32 channels x 64 cols: one uint4 global load + 8 transposed stores
      union Sv { uint4 q; unsigned short s[8]; } sv;
      sv.q = uint4{0u, 0u, 0u, 0u};
      if (rowOK)
        sv.q = *reinterpret_cast<const uint4*>(
            xb + (cbk * 32 + sch) * 4096 + row * 64 + scg * 8);
#pragma unroll
      for (int c = 0; c < 8; ++c)
        xs[(scg * 8 + c + 1) * 40 + sch] = sv.s[c];
      __syncthreads();

      // prefetch the 3 A fragments for this stage (contiguous 32B per lane)
      const int fragBase = (((b * 32 + coutBlk) * 3 + kh) * 16 + cbk) * 3;
      bf16x16 a[3];
#pragma unroll
      for (int kw = 0; kw < 3; ++kw)
        a[kw] = *reinterpret_cast<const bf16x16*>(
            wfrag + (fragBase + kw) * 512 + lane * 16);

      union Uv { uint4 q[2]; bf16x16 v; };
#pragma unroll
      for (int kw = 0; kw < 3; ++kw) {
        // batch all 4 B-fragment loads before the WMMAs
        Uv u[4];
#pragma unroll
        for (int s = 0; s < 4; ++s) {
          const int col = s * 16 + n + kw;
          u[s].q[0] = *reinterpret_cast<const uint4*>(&xs[col * 40 + kb]);
          u[s].q[1] = *reinterpret_cast<const uint4*>(&xs[col * 40 + kb + 16]);
        }
#pragma unroll
        for (int s = 0; s < 4; ++s)
          acc[s] = __builtin_amdgcn_wmma_f32_16x16x32_bf16(
              false, a[kw], false, u[s].v, (short)0, acc[s], false, false);
      }
    }
  }

  // epilogue: C/D layout -> cout = coutBlk*16 + v + (lane>=16 ? 8:0)
  const int mofs = (lane & 16) >> 1;
#pragma unroll
  for (int v = 0; v < 8; ++v) {
    const int cout = coutBlk * 16 + v + mofs;
    const float bi = bias[cout];
    const int base = ((b * COUT + cout) * H_ + h) * W_;
#pragma unroll
    for (int s = 0; s < 4; ++s)
      out[base + s * 16 + n] = lrelu_s2(acc[s][v] + bi);
  }
}

extern "C" void kernel_launch(void* const* d_in, const int* in_sizes, int n_in,
                              void* d_out, int out_size, void* d_ws, size_t ws_size,
                              hipStream_t stream) {
  (void)in_sizes; (void)n_in; (void)out_size; (void)ws_size;
  const float* x      = (const float*)d_in[0];   // [16,512,64,64]
  const float* latent = (const float*)d_in[1];   // [16,512]
  const float* noise  = (const float*)d_in[2];   // [16,1,64,64]
  const float* aw     = (const float*)d_in[3];   // [512,512]
  const float* ab     = (const float*)d_in[4];   // [512]
  const float* cw     = (const float*)d_in[5];   // [512,512,3,3]
  const float* nw     = (const float*)d_in[6];   // scalar
  const float* bias   = (const float*)d_in[7];   // [1,512,1,1]
  float* out = (float*)d_out;                    // [16,512,64,64]

  // workspace carve-up
  char* ws = (char*)d_ws;
  unsigned short* xh    = (unsigned short*)ws;                       // 67,108,864 B
  unsigned short* wfrag = (unsigned short*)(ws + 67108864);          // 75,497,472 B
  float* style = (float*)(ws + 67108864 + 75497472);                 // 32 KB
  float* demod = style + B_ * CIN;                                   // 32 KB
  float* wsq   = demod + B_ * COUT;                                  // 1 MB

  k_style<<<32,     256, 0, stream>>>(latent, aw, ab, style);
  k_wsq  <<<1024,   256, 0, stream>>>(cw, wsq);
  k_demod<<<32,     256, 0, stream>>>(style, wsq, demod);
  k_xh   <<<131072, 256, 0, stream>>>(x, noise, nw, xh);
  k_pack <<<147456, 256, 0, stream>>>(cw, style, demod, wfrag);
  k_conv <<<dim3(4, 64, 16), 256, 0, stream>>>(xh, wfrag, bias, out);
}
